// MemoryRetrievalLayer_44341242364185
// MI455X (gfx1250) — matrix operand order repaired and checked by reading, not compiled, not run on hardware
//
#include <hip/hip_runtime.h>

// ---------------- problem constants ----------------
#define NQ        512
#define ROWS      4096
#define VPR       256
#define KD        128
#define MEMDIM    128
#define KTOP      64
#define KPOST     32
#define RPB       8          // rows per workgroup in the scoring kernel
#define LARGE_NUM 1e10f

// output layout (flat floats, tuple order of the reference)
#define O_VALS   0                       // [512][32][128]
#define O_ENT    (O_VALS + NQ*KPOST*MEMDIM)
#define O_GID    (O_ENT  + NQ*KPOST)
#define O_SCORE  (O_GID  + NQ*KPOST)
#define O_MASK   (O_SCORE+ NQ*KPOST)
#define O_NDIS   (O_MASK + NQ*KPOST)
#define O_ATTN   (O_NDIS + 1)

typedef __attribute__((ext_vector_type(16))) __bf16          v16bf;
typedef __attribute__((ext_vector_type(16))) unsigned short  v16us;
typedef __attribute__((ext_vector_type(8)))  unsigned short  v8us;
typedef __attribute__((ext_vector_type(4)))  unsigned short  v4us;
typedef __attribute__((ext_vector_type(8)))  float           v8f;
typedef __attribute__((ext_vector_type(4)))  float           v4f;

static __device__ __forceinline__ unsigned short f2bf_bits(float x) {
    unsigned u = __float_as_uint(x);
    u += 0x7FFFu + ((u >> 16) & 1u);   // round-to-nearest-even
    return (unsigned short)(u >> 16);
}

// ---------------- K0: queries f32 -> bf16 ----------------
__global__ __launch_bounds__(256) void qconv_kernel(const float* __restrict__ q,
                                                    unsigned short* __restrict__ qb) {
    int i = blockIdx.x * 256 + threadIdx.x;
    if (i < NQ * KD) qb[i] = f2bf_bits(q[i]);
}

// ---------------- K1: bf16 WMMA scoring, per-row max+argmax ----------------
// Keys are streamed from HBM exactly ONCE (512 MB ~ 22us at 23.3 TB/s).
// grid = 512 row-blocks (8 rows each); block = 512 threads (16 waves).
// Each wave holds A-fragments for 2 query-blocks (32 queries) in VGPRs for
// the whole kernel; 16 waves cover all 512 queries against the shared key
// tile in LDS. Per tile: one coalesced global load + RNE convert + ds_store,
// then 4 B-fragment ds_loads feeding 8 v_wmma_f32_16x16x32_bf16.
__global__ __launch_bounds__(512) void score_rowmax_kernel(
    const unsigned short* __restrict__ Qb,     // [512][128] bf16 bits
    const float*          __restrict__ keys,   // [4096][256][128] f32
    float*                __restrict__ row_max,// [512][4096]
    int*                  __restrict__ row_arg)// [512][4096]
{
    const int tid  = threadIdx.x;
    const int wave = tid >> 5;            // 0..15
    const int lane = tid & 31;
    const int m    = lane & 15;           // A: M row / B: key-in-tile / C: N col
    const int kg   = lane >> 4;           // lane-group

    __shared__ unsigned short tile[16 * KD];  // 4 KB bf16 key tile (16 keys x 128 d)

    // ---- A-fragments for 2 query blocks, resident in VGPRs all kernel ----
    // contiguous: lanegroup0 holds K t*32+0..15, lanegroup1 t*32+16..31
    v16bf Afr[2][4];
    #pragma unroll
    for (int j = 0; j < 2; ++j) {
        const int qb = wave * 2 + j;
        const unsigned short* qrow = Qb + (qb * 16 + m) * KD;
        #pragma unroll
        for (int t = 0; t < 4; ++t) {
            v16us au = *(const v16us*)(qrow + t * 32 + kg * 16);
            Afr[j][t] = __builtin_bit_cast(v16bf, au);
        }
    }

    const int row0 = blockIdx.x * RPB;
    for (int rr = 0; rr < RPB; ++rr) {
        const int r = row0 + rr;
        const float* rowkeys = keys + (size_t)r * VPR * KD;

        float bestv[2][8];
        int   besti[2][8];
        #pragma unroll
        for (int j = 0; j < 2; ++j)
            #pragma unroll
            for (int v = 0; v < 8; ++v) { bestv[j][v] = -__builtin_inff(); besti[j][v] = 0; }

        // prologue: fetch tile 0 (4 consecutive f32 per thread, coalesced b128)
        v4f staged = *(const v4f*)(rowkeys + tid * 4);

        for (int n = 0; n < 16; ++n) {          // 16 key-tiles per row
            // ---- convert staged f32 -> bf16, store to shared tile ----
            v4us hv;
            #pragma unroll
            for (int c = 0; c < 4; ++c) hv[c] = f2bf_bits(staged[c]);
            *(v4us*)(tile + tid * 4) = hv;
            __syncthreads();

            // issue next tile's global load early; latency hides under WMMAs
            if (n < 15) staged = *(const v4f*)(rowkeys + (n + 1) * (16 * KD) + tid * 4);

            // ---- B-fragments (shared across both query blocks) ----
            v16bf Bf[4];
            #pragma unroll
            for (int t = 0; t < 4; ++t) {
                v16us bu = *(const v16us*)(tile + m * KD + t * 32 + kg * 16);
                Bf[t] = __builtin_bit_cast(v16bf, bu);
            }

            const int kidx = n * 16 + m;        // key index within row (N = lane&15)
            #pragma unroll
            for (int j = 0; j < 2; ++j) {
                v8f acc = {0.f, 0.f, 0.f, 0.f, 0.f, 0.f, 0.f, 0.f};
                #pragma unroll
                for (int t = 0; t < 4; ++t)
                    acc = __builtin_amdgcn_wmma_f32_16x16x32_bf16(
                            false, Afr[j][t], false, Bf[t], (short)0, acc, false, false);
                #pragma unroll
                for (int v = 0; v < 8; ++v) {
                    float s = acc[v];
                    if (s > bestv[j][v]) { bestv[j][v] = s; besti[j][v] = kidx; }
                }
            }
            __syncthreads();   // all reads of `tile` done before next overwrite
        }

        // ---- cross-lane reduce over the 16 lanes of each lanegroup ----
        #pragma unroll
        for (int j = 0; j < 2; ++j) {
            #pragma unroll
            for (int v = 0; v < 8; ++v) {
                float s = bestv[j][v];
                int   i = besti[j][v];
                #pragma unroll
                for (int off = 1; off < 16; off <<= 1) {
                    float so = __shfl_xor(s, off, 32);
                    int   io = __shfl_xor(i, off, 32);
                    if (so > s || (so == s && io < i)) { s = so; i = io; }
                }
                bestv[j][v] = s; besti[j][v] = i;
            }
            if (m == 0) {   // lanes 0 (M=0..7) and 16 (M=8..15) write results
                #pragma unroll
                for (int v = 0; v < 8; ++v) {
                    const int qidx = (wave * 2 + j) * 16 + v + kg * 8;
                    row_max[(size_t)qidx * ROWS + r] = bestv[j][v];
                    row_arg[(size_t)qidx * ROWS + r] = besti[j][v];
                }
            }
        }
    }
}

// ---------------- K2: top-64 rows per query ----------------
__global__ __launch_bounds__(256) void topk_kernel(
    const float* __restrict__ row_max, const int* __restrict__ row_arg,
    float* __restrict__ top_scores, int* __restrict__ top_ids)
{
    const int q = blockIdx.x;
    const int tid = threadIdx.x;
    __shared__ float sc[ROWS];
    __shared__ float rv[256];
    __shared__ int   ri[256];

    for (int i = tid; i < ROWS; i += 256) sc[i] = row_max[(size_t)q * ROWS + i];
    __syncthreads();

    for (int k = 0; k < KTOP; ++k) {
        float bv = -__builtin_inff();
        int   bi = 1 << 30;
        for (int i = tid; i < ROWS; i += 256) {
            float v = sc[i];
            if (v > bv || (v == bv && i < bi)) { bv = v; bi = i; }
        }
        rv[tid] = bv; ri[tid] = bi;
        __syncthreads();
        for (int s = 128; s > 0; s >>= 1) {
            if (tid < s) {
                float ov = rv[tid + s]; int oi = ri[tid + s];
                if (ov > rv[tid] || (ov == rv[tid] && oi < ri[tid])) { rv[tid] = ov; ri[tid] = oi; }
            }
            __syncthreads();
        }
        if (tid == 0) {
            int r = ri[0];
            top_scores[q * KTOP + k] = rv[0];
            top_ids[q * KTOP + k]    = r * VPR + row_arg[(size_t)q * ROWS + r];
            sc[r] = -__builtin_inff();
        }
        __syncthreads();
    }
}

// ---------------- K3: gather + f32 rescore + mask + softmax ----------------
__global__ __launch_bounds__(128) void finalize_kernel(
    const float* __restrict__ queries, const float* __restrict__ keys_flat,
    const float* __restrict__ mem_values, const int* __restrict__ mem_ident,
    const int* __restrict__ mem_ent, const int* __restrict__ text_ident,
    const int* __restrict__ top_ids, float* __restrict__ out,
    int* __restrict__ dis_counts)
{
    const int q = blockIdx.x;
    const int tid = threadIdx.x;
    __shared__ int   s_ids[KPOST];
    __shared__ float s_ms[KPOST];
    __shared__ int   s_mk[KPOST];

    if (tid < KPOST) s_ids[tid] = top_ids[q * KTOP + tid];   // sel == first 32 (sorted desc)
    __syncthreads();

    float* out_vals = out + O_VALS + (size_t)q * KPOST * MEMDIM;
    for (int j = 0; j < KPOST; ++j)
        out_vals[j * MEMDIM + tid] = mem_values[(size_t)s_ids[j] * MEMDIM + tid];

    if (tid < KPOST) {
        const int id = s_ids[tid];
        const float* kv = keys_flat + (size_t)id * KD;
        const float* qv = queries + (size_t)q * KD;
        float s = 0.f;
        #pragma unroll 4
        for (int d = 0; d < KD; ++d) s = fmaf(qv[d], kv[d], s);
        const int msk = (mem_ident[id] == text_ident[q]) ? 1 : 0;
        out[O_ENT   + q * KPOST + tid] = (float)mem_ent[id];
        out[O_GID   + q * KPOST + tid] = (float)id;
        out[O_SCORE + q * KPOST + tid] = s;
        out[O_MASK  + q * KPOST + tid] = (float)msk;
        s_ms[tid] = s - (float)msk * LARGE_NUM;
        s_mk[tid] = msk;
    }
    __syncthreads();

    if (tid < KPOST) {   // tid<32 == wave 0 on wave32 hardware
        float x = s_ms[tid];
        float mx = x;
        #pragma unroll
        for (int off = 1; off < 32; off <<= 1) mx = fmaxf(mx, __shfl_xor(mx, off, 32));
        float e = __expf(x - mx);
        float sum = e;
        #pragma unroll
        for (int off = 1; off < 32; off <<= 1) sum += __shfl_xor(sum, off, 32);
        out[O_ATTN + q * KPOST + tid] = e / sum;
        int c = s_mk[tid];
        #pragma unroll
        for (int off = 1; off < 32; off <<= 1) c += __shfl_xor(c, off, 32);
        if (tid == 0) dis_counts[q] = c;
    }
}

// ---------------- K4: n_disallowed = sum over queries ----------------
__global__ __launch_bounds__(256) void sum_dis_kernel(const int* __restrict__ dis_counts,
                                                      float* __restrict__ out) {
    __shared__ int s[256];
    const int tid = threadIdx.x;
    int acc = 0;
    for (int i = tid; i < NQ; i += 256) acc += dis_counts[i];
    s[tid] = acc;
    __syncthreads();
    for (int st = 128; st > 0; st >>= 1) {
        if (tid < st) s[tid] += s[tid + st];
        __syncthreads();
    }
    if (tid == 0) out[O_NDIS] = (float)s[0];
}

// ---------------- host launcher ----------------
extern "C" void kernel_launch(void* const* d_in, const int* in_sizes, int n_in,
                              void* d_out, int out_size, void* d_ws, size_t ws_size,
                              hipStream_t stream) {
    const float* queries      = (const float*)d_in[0];
    const float* memory_keys  = (const float*)d_in[1];
    const int*   mem_ident    = (const int*)d_in[2];
    const int*   mem_ent      = (const int*)d_in[3];
    const float* mem_values   = (const float*)d_in[4];
    const int*   text_ident   = (const int*)d_in[5];
    float* out = (float*)d_out;

    // workspace carve (~16.5 MB)
    char* w = (char*)d_ws;
    unsigned short* Qb     = (unsigned short*)w; w += (size_t)NQ * KD * 2;
    float* row_max         = (float*)w;          w += (size_t)NQ * ROWS * 4;
    int*   row_arg         = (int*)w;            w += (size_t)NQ * ROWS * 4;
    float* top_scores      = (float*)w;          w += (size_t)NQ * KTOP * 4;
    int*   top_ids         = (int*)w;            w += (size_t)NQ * KTOP * 4;
    int*   dis_counts      = (int*)w;            w += (size_t)NQ * 4;

    qconv_kernel<<<(NQ * KD + 255) / 256, 256, 0, stream>>>(queries, Qb);

    score_rowmax_kernel<<<ROWS / RPB, 512, 0, stream>>>(Qb, memory_keys, row_max, row_arg);

    topk_kernel<<<NQ, 256, 0, stream>>>(row_max, row_arg, top_scores, top_ids);

    finalize_kernel<<<NQ, 128, 0, stream>>>(queries, memory_keys, mem_values,
                                            mem_ident, mem_ent, text_ident,
                                            top_ids, out, dis_counts);

    sum_dis_kernel<<<1, 256, 0, stream>>>(dis_counts, out);
}
